// SwinTransformer_86260123172926
// MI455X (gfx1250) — matrix operand order repaired and checked by reading, not compiled, not run on hardware
//
#include <hip/hip_runtime.h>
#include <hip/hip_bf16.h>

// ---------------- static config (mirrors reference) ----------------
#define IMG   56
#define WIN   7
#define SHIFT 3
#define CCH   256
#define NH    8
#define DH    32
#define SW    49      // tokens per window
#define NYW   8
#define NWIN  64      // windows per image
#define BATCH 32
#define NTOK  (IMG*IMG)               // 3136
#define MROWS (BATCH*NTOK)            // 100352 rows of the big GEMMs
#define NMLP  1024
#define EPS_LN 1e-5f
#define ATT_SCALE 0.17677669529663687f  // 32^-0.5

typedef __attribute__((ext_vector_type(16))) __bf16 v16bf;
typedef __attribute__((ext_vector_type(8)))  __bf16 bf16x8;
typedef __attribute__((ext_vector_type(4)))  __bf16 bf16x4;
typedef __attribute__((ext_vector_type(8)))  float  v8f;

// ---------------- WMMA helper ----------------
__device__ inline v8f wmma_bf16(v16bf a, v16bf b, v8f c) {
  // D = A(16x32 bf16) * B(32x16 bf16) + C(16x16 f32)
  return __builtin_amdgcn_wmma_f32_16x16x32_bf16(
      false, a, false, b, (short)0, c, false, false);
}

// ---------------- CDNA5 async global->LDS (ASYNCcnt path) ----------------
// GLOBAL_LOAD_ASYNC_TO_LDS_B128 (ISA §15.18.3 op 98): per-lane copy of 16B
// from global memory directly into LDS, tracked by ASYNCcnt.  vdst holds the
// 32-bit LDS byte offset (generic-pointer low 32 bits, per ISA §10.2).
__device__ inline void async_load_b128(unsigned lds_off, const void* gptr) {
  asm volatile("global_load_async_to_lds_b128 %0, %1, off"
               :: "v"(lds_off), "v"(gptr) : "memory");
}
__device__ inline void wait_async0() {
  asm volatile("s_wait_asynccnt 0x0" ::: "memory");
}
__device__ inline unsigned lds_off32(const void* p) {
  return (unsigned)(unsigned long long)p;
}

// A-fragment (16x32 bf16, M x K) from LDS, row-major with leading dim `ld`
// (halfword units).  Per ISA 7.12.2: lanes 0-15 hold row M=lane with
// K = {0..7, 16..23}; lanes 16-31 hold row M=lane-16 with K = {8..15, 24..31}.
// Each half is 8 contiguous halfwords -> one b128 LDS read.
__device__ inline v16bf load_fragA(const __bf16* lds, int row0, int ld) {
  int lane = threadIdx.x & 31;
  int r  = row0 + (lane & 15);
  int k0 = (lane < 16) ? 0 : 8;
  union { v16bf v; bf16x8 h[2]; } u;
  u.h[0] = *(const bf16x8*)(lds + r * ld + k0);
  u.h[1] = *(const bf16x8*)(lds + r * ld + k0 + 16);
  return u.v;
}

// B-fragment (32x16 bf16, K x N) from LDS holding B *transposed* ([N][K],
// leading dim `ld`).  Lanes 0-15: column N=lane, K={0..7, 16..23}; lanes 16-31:
// column N=lane-16, K={8..15, 24..31} (two 8-halfword runs per lane).
__device__ inline v16bf load_fragB(const __bf16* ldsT, int col0, int ld) {
  int lane = threadIdx.x & 31;
  int c  = col0 + (lane & 15);
  int k0 = (lane < 16) ? 0 : 8;
  union { v16bf v; bf16x8 h[2]; } u;
  u.h[0] = *(const bf16x8*)(ldsT + c * ld + k0);
  u.h[1] = *(const bf16x8*)(ldsT + c * ld + k0 + 16);
  return u.v;
}

// ---------------- fp32 -> bf16 weight conversion ----------------
__global__ void cvt_kernel(const float* __restrict__ s, __bf16* __restrict__ d, int n) {
  int i = blockIdx.x * blockDim.x + threadIdx.x;
  if (i < n) d[i] = (__bf16)s[i];
}

// ---------------- LayerNorm (+ optional shift & window partition) ----------------
// mode 0: dst row rw is window-order; source token = roll(-3,-3) pixel.
// mode 1: identity mapping (second LayerNorm).
__global__ __launch_bounds__(256) void ln_kernel(
    const float* __restrict__ src, const float* __restrict__ gamma,
    const float* __restrict__ beta, __bf16* __restrict__ dst, int mode) {
  int gwave = (blockIdx.x * blockDim.x + threadIdx.x) >> 5;
  int lane  = threadIdx.x & 31;
  if (gwave >= MROWS) return;

  int srow;
  if (mode == 0) {
    int b = gwave / NTOK, r = gwave % NTOK;
    int w = r / SW, s = r % SW;
    int wy = w >> 3, wx = w & 7, i = s / WIN, j = s % WIN;
    int p = wy * WIN + i, q = wx * WIN + j;           // shifted coords
    int sp = (p + SHIFT) % IMG, sq = (q + SHIFT) % IMG;
    srow = b * NTOK + sp * IMG + sq;
  } else {
    srow = gwave;
  }

  const float4* xp = (const float4*)(src + (size_t)srow * CCH);
  float4 a = xp[lane];
  float4 b = xp[lane + 32];
  float s = a.x + a.y + a.z + a.w + b.x + b.y + b.z + b.w;
  #pragma unroll
  for (int m = 1; m < 32; m <<= 1) s += __shfl_xor(s, m);
  float mu = s * (1.0f / CCH);
  float v =
      (a.x-mu)*(a.x-mu) + (a.y-mu)*(a.y-mu) + (a.z-mu)*(a.z-mu) + (a.w-mu)*(a.w-mu) +
      (b.x-mu)*(b.x-mu) + (b.y-mu)*(b.y-mu) + (b.z-mu)*(b.z-mu) + (b.w-mu)*(b.w-mu);
  #pragma unroll
  for (int m = 1; m < 32; m <<= 1) v += __shfl_xor(v, m);
  float rs = rsqrtf(v * (1.0f / CCH) + EPS_LN);

  const float4* gp = (const float4*)gamma;
  const float4* bp = (const float4*)beta;
  float4 g0 = gp[lane], g1 = gp[lane + 32];
  float4 e0 = bp[lane], e1 = bp[lane + 32];
  bf16x4 o0, o1;
  o0[0]=(__bf16)((a.x-mu)*rs*g0.x+e0.x); o0[1]=(__bf16)((a.y-mu)*rs*g0.y+e0.y);
  o0[2]=(__bf16)((a.z-mu)*rs*g0.z+e0.z); o0[3]=(__bf16)((a.w-mu)*rs*g0.w+e0.w);
  o1[0]=(__bf16)((b.x-mu)*rs*g1.x+e1.x); o1[1]=(__bf16)((b.y-mu)*rs*g1.y+e1.y);
  o1[2]=(__bf16)((b.z-mu)*rs*g1.z+e1.z); o1[3]=(__bf16)((b.w-mu)*rs*g1.w+e1.w);
  *(bf16x4*)(dst + (size_t)gwave * CCH + lane * 4)       = o0;
  *(bf16x4*)(dst + (size_t)gwave * CCH + 128 + lane * 4) = o1;
}

// ---------------- generic bf16 WMMA GEMM with fused epilogues ----------------
// C[M,N] = A[M,K] @ W[K,N] + bias ; mode selects epilogue.
#define LDT 40   // LDS leading dim (halfwords), 80B rows: 16B aligned + bank skew
__global__ __launch_bounds__(256) void gemm_kernel(
    const __bf16* __restrict__ A, const __bf16* __restrict__ W,
    const float* __restrict__ bias, int M, int N, int K, int mode,
    __bf16* __restrict__ outQ, __bf16* __restrict__ outK, __bf16* __restrict__ outV,
    const float* __restrict__ resid, float* __restrict__ outF,
    __bf16* __restrict__ outH) {
  __shared__ __bf16 sA[128 * LDT];
  __shared__ __bf16 sB[128 * LDT];   // transposed: [n][k]

  int tid = threadIdx.x, lane = tid & 31, wave = tid >> 5;
  int wm = wave >> 2, wn = wave & 3;          // 2 x 4 wave grid
  int mblk = blockIdx.y * 128, nblk = blockIdx.x * 128;

  v8f acc[4][2] = {};

  for (int kb = 0; kb < K; kb += 32) {
    { // stage A tile 128x32 row-major via async global->LDS (ASYNCcnt)
      int row = tid >> 1, seg = (tid & 1) * 16;
      const __bf16* g = A + (size_t)(mblk + row) * K + kb + seg;
      unsigned l = lds_off32(sA + row * LDT + seg);
      async_load_b128(l,      g);
      async_load_b128(l + 16, g + 8);
    }
    { // stage B tile 32x128, transposed into [n][k]
      int k = tid >> 3, n0 = (tid & 7) * 16;
      const bf16x8* g = (const bf16x8*)(W + (size_t)(kb + k) * N + nblk + n0);
      bf16x8 x0 = g[0], x1 = g[1];
      #pragma unroll
      for (int i = 0; i < 8; ++i) sB[(n0 + i) * LDT + k] = x0[i];
      #pragma unroll
      for (int i = 0; i < 8; ++i) sB[(n0 + 8 + i) * LDT + k] = x1[i];
    }
    wait_async0();
    __syncthreads();

    if (kb + 32 < K) { // prefetch next K-tile (global_prefetch_b8)
      __builtin_prefetch(A + (size_t)(mblk + (tid >> 1)) * K + kb + 32, 0, 1);
      __builtin_prefetch(W + (size_t)(kb + 32 + (tid >> 3)) * N + nblk, 0, 1);
    }

    v16bf bf0 = load_fragB(sB, wn * 32,      LDT);
    v16bf bf1 = load_fragB(sB, wn * 32 + 16, LDT);
    #pragma unroll
    for (int mt = 0; mt < 4; ++mt) {
      v16bf af = load_fragA(sA, wm * 64 + mt * 16, LDT);
      acc[mt][0] = wmma_bf16(af, bf0, acc[mt][0]);
      acc[mt][1] = wmma_bf16(af, bf1, acc[mt][1]);
    }
    __syncthreads();
  }

  // epilogue.  C-layout: element e of v8f -> row = e + (lane>=16 ? 8 : 0),
  // col = lane & 15 (ISA 7.12.2).
  int l15 = lane & 15, hi8 = (lane >= 16) ? 8 : 0;
  #pragma unroll
  for (int mt = 0; mt < 4; ++mt) {
    #pragma unroll
    for (int nt = 0; nt < 2; ++nt) {
      #pragma unroll
      for (int e = 0; e < 8; ++e) {
        int row = mblk + wm * 64 + mt * 16 + e + hi8;
        int col = nblk + wn * 32 + nt * 16 + l15;
        float v = acc[mt][nt][e] + bias[col];
        if (mode == 0) {           // QKV: split, scale q, pack [w,h,s,d] bf16
          int w = row / SW, s = row % SW;
          int which = col >> 8, hh = (col & 255) >> 5, d = col & 31;
          if (which == 0) v *= ATT_SCALE;
          __bf16* t = (which == 0) ? outQ : (which == 1) ? outK : outV;
          t[(((size_t)w * NH + hh) * SW + s) * DH + d] = (__bf16)v;
        } else if (mode == 1) {    // proj: window reverse + unshift + residual
          int b = row / NTOK, r = row % NTOK;
          int w = r / SW, s = r % SW;
          int wy = w >> 3, wx = w & 7, i = s / WIN, j = s % WIN;
          int p = wy * WIN + i, q = wx * WIN + j;
          int tok = b * NTOK + ((p + SHIFT) % IMG) * IMG + ((q + SHIFT) % IMG);
          size_t o = (size_t)tok * CCH + col;
          outF[o] = resid[o] + v;
        } else if (mode == 2) {    // MLP1: exact GELU -> bf16 hidden
          float g = 0.5f * v * (1.0f + erff(v * 0.70710678118654752f));
          outH[(size_t)row * N + col] = (__bf16)g;
        } else {                   // MLP2: + residual -> f32 out
          size_t o = (size_t)row * N + col;
          outF[o] = resid[o] + v;
        }
      }
    }
  }
}

// ---------------- attention: one wave per (window, head) ----------------
__device__ inline int win_region(int p, int q) {
  // faithful to reference mask construction (cols 49..52 stay region 0)
  int rh = (p < 49) ? 0 : ((p < 53) ? 1 : 2);
  return (q < 49) ? rh * 3 : ((q >= 53) ? rh * 3 + 2 : 0);
}

__global__ __launch_bounds__(128) void attn_kernel(
    const __bf16* __restrict__ Q, const __bf16* __restrict__ Kb,
    const __bf16* __restrict__ V, const float* __restrict__ relt,
    __bf16* __restrict__ out) {
  // per-wave LDS: q[64*32] k[64*32] (later overlaid by probs[64*64]) vt[32*64]
  __shared__ __bf16 smem[4 * 6144];
  int lane = threadIdx.x & 31, wave = threadIdx.x >> 5;
  int gid = blockIdx.x * 4 + wave;          // (w,h) pair id, 0..16383
  int w = gid >> 3, h = gid & 7;
  int wimg = w & (NWIN - 1);
  int wy = wimg >> 3, wx = wimg & 7;

  __bf16* q     = smem + wave * 6144;
  __bf16* k     = q + 2048;
  __bf16* vt    = q + 4096;                 // [32 d][64 kpos]
  __bf16* probs = q;                        // overlays q+k after scores read

  const __bf16* qg = Q  + (size_t)gid * SW * DH;
  const __bf16* kg = Kb + (size_t)gid * SW * DH;
  const __bf16* vg = V  + (size_t)gid * SW * DH;

  bf16x8 z;
  #pragma unroll
  for (int i = 0; i < 8; ++i) z[i] = (__bf16)0.0f;

  for (int r = lane; r < 64; r += 32) {
    if (r < SW) {
      // async global->LDS staging for q/k rows (4 x 16B each)
      unsigned lq = lds_off32(q + r * 32);
      unsigned lk = lds_off32(k + r * 32);
      const __bf16* sq = qg + r * 32;
      const __bf16* sk = kg + r * 32;
      async_load_b128(lq,      sq);
      async_load_b128(lq + 16, sq + 8);
      async_load_b128(lq + 32, sq + 16);
      async_load_b128(lq + 48, sq + 24);
      async_load_b128(lk,      sk);
      async_load_b128(lk + 16, sk + 8);
      async_load_b128(lk + 32, sk + 16);
      async_load_b128(lk + 48, sk + 24);
      const __bf16* vr = vg + r * 32;
      #pragma unroll
      for (int d = 0; d < 32; ++d) vt[d * 64 + r] = vr[d];
    } else {
      bf16x8* dq = (bf16x8*)(q + r * 32);
      bf16x8* dk = (bf16x8*)(k + r * 32);
      dq[0]=z; dq[1]=z; dq[2]=z; dq[3]=z;
      dk[0]=z; dk[1]=z; dk[2]=z; dk[3]=z;
      #pragma unroll
      for (int d = 0; d < 32; ++d) vt[d * 64 + r] = (__bf16)0.0f;
    }
  }
  wait_async0();
  // wave-private LDS region: DS ops are in-order per wave, no barrier needed.

  // scores = q @ k^T  (64x64 padded, K=32): for B, column n is key-row n,
  // so k row-major [64][32] is exactly the transposed-B layout.
  v8f sc[4][4] = {};
  v16bf kf[4];
  #pragma unroll
  for (int nt = 0; nt < 4; ++nt) kf[nt] = load_fragB(k, nt * 16, 32);
  #pragma unroll
  for (int mt = 0; mt < 4; ++mt) {
    v16bf af = load_fragA(q, mt * 16, 32);
    #pragma unroll
    for (int nt = 0; nt < 4; ++nt) sc[mt][nt] = wmma_bf16(af, kf[nt], sc[mt][nt]);
  }

  int l15 = lane & 15, hi8 = (lane >= 16) ? 8 : 0;

  // bias + shift-mask (on the fly), pad cols -> -inf
  #pragma unroll
  for (int nt = 0; nt < 4; ++nt) {
    int col = nt * 16 + l15;
    int it = col / WIN, jt = col % WIN;
    int regT = win_region(wy * WIN + it, wx * WIN + jt);
    #pragma unroll
    for (int mt = 0; mt < 4; ++mt) {
      #pragma unroll
      for (int e = 0; e < 8; ++e) {
        int row = mt * 16 + e + hi8;
        if (col >= SW) {
          sc[mt][nt][e] = -1e30f;
        } else if (row < SW) {
          int is = row / WIN, js = row % WIN;
          int regS = win_region(wy * WIN + is, wx * WIN + js);
          int idx = (is - it + WIN - 1) * (2 * WIN - 1) + (js - jt + WIN - 1);
          float b = relt[idx * NH + h] + ((regS != regT) ? -100.0f : 0.0f);
          sc[mt][nt][e] += b;
        }
      }
    }
  }

  // softmax per row (row lives on a fixed 16-lane half; reduce with xor<16)
  #pragma unroll
  for (int mt = 0; mt < 4; ++mt) {
    #pragma unroll
    for (int e = 0; e < 8; ++e) {
      float m = sc[mt][0][e];
      m = fmaxf(m, sc[mt][1][e]); m = fmaxf(m, sc[mt][2][e]); m = fmaxf(m, sc[mt][3][e]);
      #pragma unroll
      for (int msk = 1; msk < 16; msk <<= 1) m = fmaxf(m, __shfl_xor(m, msk));
      float p0 = __expf(sc[mt][0][e] - m), p1 = __expf(sc[mt][1][e] - m);
      float p2 = __expf(sc[mt][2][e] - m), p3 = __expf(sc[mt][3][e] - m);
      float s = p0 + p1 + p2 + p3;
      #pragma unroll
      for (int msk = 1; msk < 16; msk <<= 1) s += __shfl_xor(s, msk);
      float inv = 1.0f / s;
      int row = mt * 16 + e + hi8;
      probs[row * 64 +      l15] = (__bf16)(p0 * inv);
      probs[row * 64 + 16 + l15] = (__bf16)(p1 * inv);
      probs[row * 64 + 32 + l15] = (__bf16)(p2 * inv);
      probs[row * 64 + 48 + l15] = (__bf16)(p3 * inv);
    }
  }

  // out = P(64x64) @ V(64x32), two K-chunks of 32
  v8f o[4][2] = {};
  #pragma unroll
  for (int kk = 0; kk < 2; ++kk) {
    v16bf vb0 = load_fragB(vt + kk * 32, 0,  64);
    v16bf vb1 = load_fragB(vt + kk * 32, 16, 64);
    #pragma unroll
    for (int mt = 0; mt < 4; ++mt) {
      v16bf af = load_fragA(probs + kk * 32, mt * 16, 64);
      o[mt][0] = wmma_bf16(af, vb0, o[mt][0]);
      o[mt][1] = wmma_bf16(af, vb1, o[mt][1]);
    }
  }

  #pragma unroll
  for (int mt = 0; mt < 4; ++mt) {
    #pragma unroll
    for (int nt = 0; nt < 2; ++nt) {
      #pragma unroll
      for (int e = 0; e < 8; ++e) {
        int row = mt * 16 + e + hi8;
        if (row < SW)
          out[((size_t)w * SW + row) * CCH + h * DH + nt * 16 + l15] =
              (__bf16)o[mt][nt][e];
      }
    }
  }
}

// ---------------- host orchestration ----------------
extern "C" void kernel_launch(void* const* d_in, const int* in_sizes, int n_in,
                              void* d_out, int out_size, void* d_ws, size_t ws_size,
                              hipStream_t stream) {
  (void)in_sizes; (void)n_in; (void)out_size; (void)ws_size;
  const float* x      = (const float*)d_in[0];
  const float* gamma  = (const float*)d_in[1];
  const float* beta   = (const float*)d_in[2];
  const float* qkv_w  = (const float*)d_in[3];
  const float* qkv_b  = (const float*)d_in[4];
  const float* proj_w = (const float*)d_in[5];
  const float* proj_b = (const float*)d_in[6];
  const float* reltab = (const float*)d_in[7];
  const float* mlp_w1 = (const float*)d_in[8];
  const float* mlp_b1 = (const float*)d_in[9];
  const float* mlp_w2 = (const float*)d_in[10];
  const float* mlp_b2 = (const float*)d_in[11];
  float* out = (float*)d_out;

  // workspace layout (bytes). SZ = one bf16 [100352,256] buffer.
  const size_t SZ = (size_t)MROWS * CCH * 2;   // 51,380,224
  char* ws = (char*)d_ws;
  __bf16* xw   = (__bf16*)ws;                  // LN1 output, window order
  __bf16* qb   = (__bf16*)(ws + SZ);
  __bf16* kb   = (__bf16*)(ws + 2 * SZ);
  __bf16* vb   = (__bf16*)(ws + 3 * SZ);
  __bf16* hid  = (__bf16*)ws;                  // MLP hidden aliases xw+qkv (dead)
  __bf16* att  = (__bf16*)(ws + 4 * SZ);       // attention output
  float*  x1   = (float*)(ws + 5 * SZ);        // post-attention residual (f32)
  __bf16* ln2  = (__bf16*)(ws + 7 * SZ);
  __bf16* wqkv = (__bf16*)(ws + 8 * SZ);
  __bf16* wprj = wqkv + 256 * 768;
  __bf16* w1b  = wprj + 256 * 256;
  __bf16* w2b  = w1b + 256 * 1024;

  // 1) weights -> bf16
  cvt_kernel<<<(256*768 + 255)/256, 256, 0, stream>>>(qkv_w, wqkv, 256*768);
  cvt_kernel<<<(256*256 + 255)/256, 256, 0, stream>>>(proj_w, wprj, 256*256);
  cvt_kernel<<<(256*1024 + 255)/256, 256, 0, stream>>>(mlp_w1, w1b, 256*1024);
  cvt_kernel<<<(1024*256 + 255)/256, 256, 0, stream>>>(mlp_w2, w2b, 1024*256);

  // 2) LN + shift + window partition
  ln_kernel<<<MROWS/8, 256, 0, stream>>>(x, gamma, beta, xw, 0);

  // 3) QKV GEMM (M=100352, N=768, K=256)
  gemm_kernel<<<dim3(768/128, MROWS/128), 256, 0, stream>>>(
      xw, wqkv, qkv_b, MROWS, 768, 256, 0, qb, kb, vb, nullptr, nullptr, nullptr);

  // 4) windowed attention (16384 wave-jobs, 4 per block)
  attn_kernel<<<16384/4, 128, 0, stream>>>(qb, kb, vb, reltab, att);

  // 5) proj GEMM + window reverse + unshift + residual -> x1
  gemm_kernel<<<dim3(256/128, MROWS/128), 256, 0, stream>>>(
      att, wprj, proj_b, MROWS, 256, 256, 1, nullptr, nullptr, nullptr, x, x1, nullptr);

  // 6) LN2 (same gamma/beta, identity mapping)
  ln_kernel<<<MROWS/8, 256, 0, stream>>>(x1, gamma, beta, ln2, 1);

  // 7) MLP1 + GELU -> hidden (bf16)
  gemm_kernel<<<dim3(1024/128, MROWS/128), 256, 0, stream>>>(
      ln2, w1b, mlp_b1, MROWS, 1024, 256, 2, nullptr, nullptr, nullptr,
      nullptr, nullptr, hid);

  // 8) MLP2 + residual -> d_out (f32)
  gemm_kernel<<<dim3(256/128, MROWS/128), 256, 0, stream>>>(
      hid, w2b, mlp_b2, MROWS, 256, 1024, 3, nullptr, nullptr, nullptr,
      x1, out, nullptr);
}
